// CosinSimAttention_76656576299359
// MI455X (gfx1250) — matrix operand order repaired and checked by reading, not compiled, not run on hardware
//
#include <hip/hip_runtime.h>
#include <stdint.h>

typedef __attribute__((ext_vector_type(16))) _Float16 v16h;
typedef __attribute__((ext_vector_type(8)))  _Float16 v8h;
typedef __attribute__((ext_vector_type(8)))  float    v8f;
typedef __attribute__((ext_vector_type(4)))  float    v4f;

#define NB 32
#define SL 2048
#define HD 64
#define QT 16                       // queries per workgroup
#define NKT (SL / 16)               // 128 key tiles
#define WAVES 8
#define TPW (NKT / WAVES)           // 16 key tiles per wave

// ---------------------------------------------------------------------------
// Prep kernel: once per call, produce
//   qn, kn : L2-normalized f16, row-major [B][L][D]
//   vt     : f16 transposed [B][D][L]
// ---------------------------------------------------------------------------
__global__ __launch_bounds__(256)
void cosattn_prep_kernel(const float* __restrict__ q,
                         const float* __restrict__ k,
                         const float* __restrict__ v,
                         _Float16* __restrict__ qn,
                         _Float16* __restrict__ kn,
                         _Float16* __restrict__ vt)
{
    __shared__ __align__(16) _Float16 ldsv[64 * 64];   // 8 KB transpose tile

    const int tid  = threadIdx.x;
    const int b    = blockIdx.x / (SL / 64);
    const int r0   = (blockIdx.x % (SL / 64)) * 64;
    const int rloc = tid >> 2;                // 0..63 row within tile
    const int row  = r0 + rloc;
    const int dseg = (tid & 3) * 16;          // 4 threads cover one row
    const size_t base = ((size_t)b * SL + row) * HD + dseg;

    // ---- normalize q and k rows, store f16 row-major ----
#pragma unroll
    for (int which = 0; which < 2; ++which) {
        const float* src = which ? k : q;
        _Float16*    dst = which ? kn : qn;
        float xv[16];
        *(v4f*)&xv[0]  = __builtin_nontemporal_load((const v4f*)(src + base));
        *(v4f*)&xv[4]  = __builtin_nontemporal_load((const v4f*)(src + base + 4));
        *(v4f*)&xv[8]  = __builtin_nontemporal_load((const v4f*)(src + base + 8));
        *(v4f*)&xv[12] = __builtin_nontemporal_load((const v4f*)(src + base + 12));
        float ss = 0.0f;
#pragma unroll
        for (int i = 0; i < 16; ++i) ss += xv[i] * xv[i];
        ss += __shfl_xor(ss, 1, 32);          // reduce over the 4 threads of a row
        ss += __shfl_xor(ss, 2, 32);
        const float inv = 1.0f / fmaxf(sqrtf(ss), 1e-12f);
        v16h h;
#pragma unroll
        for (int i = 0; i < 16; ++i) h[i] = (_Float16)(xv[i] * inv);
        *(v16h*)(dst + base) = h;             // keep in L2 (re-read by main kernel)
    }

    // ---- v: convert to f16 and transpose via LDS ----
    {
        float xv[16];
        *(v4f*)&xv[0]  = __builtin_nontemporal_load((const v4f*)(v + base));
        *(v4f*)&xv[4]  = __builtin_nontemporal_load((const v4f*)(v + base + 4));
        *(v4f*)&xv[8]  = __builtin_nontemporal_load((const v4f*)(v + base + 8));
        *(v4f*)&xv[12] = __builtin_nontemporal_load((const v4f*)(v + base + 12));
#pragma unroll
        for (int i = 0; i < 16; ++i)
            ldsv[(dseg + i) * 64 + rloc] = (_Float16)xv[i];
    }
    __syncthreads();
    {
        const int d      = tid >> 2;          // 0..63
        const int keyseg = (tid & 3) * 16;    // 0,16,32,48
        v16h h = *(const v16h*)&ldsv[d * 64 + keyseg];
        *(v16h*)(vt + ((size_t)b * HD + d) * SL + (size_t)(r0 + keyseg)) = h;
    }
}

// ---------------------------------------------------------------------------
// Main fused kernel: one WG per (batch, 16-query tile); 8 waves.
// ---------------------------------------------------------------------------
__global__ __launch_bounds__(256)
void cosattn_main_kernel(const _Float16* __restrict__ qn,
                         const _Float16* __restrict__ kn,
                         const _Float16* __restrict__ vt,
                         const uint8_t*  __restrict__ mask,
                         float* __restrict__ out,    // [B][L][D]
                         float* __restrict__ attn)   // [B][L][L]
{
    __shared__ __align__(16) _Float16 lds_st[WAVES][2][16 * 16]; // 8 KB f16 staging
    __shared__ uint64_t lds_mask[WAVES][32];                     // 2 KB mask tiles
    __shared__ float lds_rowsum[QT];
    __shared__ __align__(16) float lds_out[QT * HD];             // 4 KB

    const int tid  = threadIdx.x;
    const int wv   = tid >> 5;
    const int lane = tid & 31;
    const int hi   = (lane >= 16) ? 1 : 0;
    const int ln16 = lane & 15;

    const int bq = blockIdx.x;                // 0 .. NB*(SL/QT)-1
    const int b  = bq / (SL / QT);
    const int q0 = (bq % (SL / QT)) * QT;

    const size_t qk_batch  = (size_t)b * SL * HD;
    const size_t mask_row0 = ((size_t)b * SL + q0) * (size_t)SL;

    if (tid < QT) lds_rowsum[tid] = 0.0f;
#pragma unroll
    for (int i = tid; i < QT * HD; i += 256) lds_out[i] = 0.0f;
    __syncthreads();

    // ---- hoisted per-thread base pointers (all loop offsets become literals) ----
    const _Float16* kbase = kn + qk_batch + (size_t)(wv * 16 + ln16) * HD + (hi ? 16 : 0);
    const uint8_t*  mbase = mask + mask_row0 + (size_t)(lane >> 1) * SL
                                 + (size_t)((lane & 1) * 8) + (size_t)(wv * 16);
    const uint8_t*  mb    = (const uint8_t*)lds_mask[wv] + hi * 128 + ln16;
    float*          abase = attn + mask_row0 + (size_t)(hi * 8) * SL
                                 + (size_t)ln16 + (size_t)(wv * 16);
    const _Float16* vbase = vt + ((size_t)b * HD + ln16) * SL
                                 + (size_t)(wv * 16) + (size_t)(hi * 128);
    _Float16*       stb   = &lds_st[wv][0][hi * 128 + ln16];

    // ---- q A-fragments: pure vector loads (layout-matched strips) ----
    v16h aq0, aq1;
    {
        const _Float16* qp = qn + qk_batch + (size_t)(q0 + ln16) * HD + (hi ? 16 : 0);
        v8h a0lo = *(const v8h*)(qp);
        v8h a0hi = *(const v8h*)(qp + 8);
        v8h a1lo = *(const v8h*)(qp + 32);
        v8h a1hi = *(const v8h*)(qp + 40);
        aq0 = __builtin_shufflevector(a0lo, a0hi, 0,1,2,3,4,5,6,7,8,9,10,11,12,13,14,15);
        aq1 = __builtin_shufflevector(a1lo, a1hi, 0,1,2,3,4,5,6,7,8,9,10,11,12,13,14,15);
    }

    // ---------------- phase 1: S = qn*kn^T + 1, masked ----------------
    float accS[TPW][8];
    float rs[8];
#pragma unroll
    for (int p = 0; p < 8; ++p) rs[p] = 0.0f;

#pragma unroll
    for (int it = 0; it < TPW; ++it) {
        // B fragments: contiguous d-strips, literal offsets (16 KB per iteration)
        v16h b0 = *(const v16h*)(kbase + it * (WAVES * 16 * HD));
        v16h b1 = *(const v16h*)(kbase + it * (WAVES * 16 * HD) + 32);

        // mask tile 16x16 bytes: one coalesced NT b64 per lane -> LDS
        lds_mask[wv][lane] =
            __builtin_nontemporal_load((const uint64_t*)(mbase + it * 128));
        __builtin_amdgcn_wave_barrier();   // DS in-order within wave

        v8f c = {0.f, 0.f, 0.f, 0.f, 0.f, 0.f, 0.f, 0.f};
        c = __builtin_amdgcn_wmma_f32_16x16x32_f16(false, aq0, false, b0, (short)0, c, false, false);
        c = __builtin_amdgcn_wmma_f32_16x16x32_f16(false, aq1, false, b1, (short)0, c, false, false);

#pragma unroll
        for (int p = 0; p < 8; ++p) {
            float val = mb[p * 16] ? 0.0f : (c[p] + 1.0f);
            accS[it][p] = val;
            rs[p] += val;
        }
        __builtin_amdgcn_wave_barrier();   // keep next iter's mask store after reads
    }

    // ---------------- phase 2: row-sum reduction ----------------
#pragma unroll
    for (int p = 0; p < 8; ++p) {
        float r = rs[p];
        r += __shfl_xor(r, 1, 32);
        r += __shfl_xor(r, 2, 32);
        r += __shfl_xor(r, 4, 32);
        r += __shfl_xor(r, 8, 32);   // reduced within each 16-lane group
        rs[p] = r;
    }
    if (ln16 == 0) {
#pragma unroll
        for (int p = 0; p < 8; ++p)
            atomicAdd(&lds_rowsum[p + hi * 8], rs[p]);
    }
    __syncthreads();
    float invr[8];
#pragma unroll
    for (int p = 0; p < 8; ++p)
        invr[p] = 1.0f / fmaxf(lds_rowsum[p + hi * 8], 1e-12f);

    // ---------------- phase 3: write attn (NT); out += attn * v ----------------
    v8f oacc[4];
#pragma unroll
    for (int nt = 0; nt < 4; ++nt)
        oacc[nt] = (v8f){0.f, 0.f, 0.f, 0.f, 0.f, 0.f, 0.f, 0.f};

#pragma unroll
    for (int j = 0; j < TPW / 2; ++j) {
        // normalize, emit attn (f32, streaming) to HBM, stage f16 for transpose
#pragma unroll
        for (int t = 0; t < 2; ++t) {
            const int it = 2 * j + t;
#pragma unroll
            for (int p = 0; p < 8; ++p) {
                float val = accS[it][p] * invr[p];
                __builtin_nontemporal_store(val, abase + (size_t)p * SL + j * 256 + t * 128);
                stb[t * 256 + p * 16] = (_Float16)val;
            }
        }
        __builtin_amdgcn_wave_barrier();

        // A fragment (attn 16x32): two contiguous f16 strips from staging
        v16h af;
        {
            const int kb2 = hi ? 8 : 0;
            v8h lo = *(const v8h*)&lds_st[wv][hi][ln16 * 16];
            v8h h8 = *(const v8h*)&lds_st[wv][hi][ln16 * 16 + 8];
            af = __builtin_shufflevector(lo, h8, 0,1,2,3,4,5,6,7,8,9,10,11,12,13,14,15);
        }
        __builtin_amdgcn_wave_barrier();

        // B fragments from transposed f16 v: contiguous key-strip, literal offsets
#pragma unroll
        for (int nt = 0; nt < 4; ++nt) {
            v16h bf = *(const v16h*)(vbase + (size_t)nt * 16 * SL + j * 256);
            oacc[nt] = __builtin_amdgcn_wmma_f32_16x16x32_f16(false, af, false, bf, (short)0, oacc[nt], false, false);
        }
    }

    // ---------------- phase 4: cross-wave reduce + store output ----------------
    {
        float* ob = lds_out + hi * 8 * HD + ln16;
#pragma unroll
        for (int nt = 0; nt < 4; ++nt) {
#pragma unroll
            for (int p = 0; p < 8; ++p)
                atomicAdd(ob + p * HD + nt * 16, oacc[nt][p]);
        }
    }
    __syncthreads();
    ((float4*)(out + qk_batch + (size_t)q0 * HD))[tid] = ((float4*)lds_out)[tid];
}

extern "C" void kernel_launch(void* const* d_in, const int* in_sizes, int n_in,
                              void* d_out, int out_size, void* d_ws, size_t ws_size,
                              hipStream_t stream) {
    const float*   q    = (const float*)d_in[0];
    const float*   k    = (const float*)d_in[1];
    const float*   v    = (const float*)d_in[2];
    const uint8_t* mask = (const uint8_t*)d_in[3];

    float* out  = (float*)d_out;                       // output first in tuple
    float* attn = out + (size_t)NB * SL * HD;          // then attn

    // workspace: qn | kn | vt, each NB*SL*HD f16 (8 MB each, 24 MB total)
    _Float16* qn = (_Float16*)d_ws;
    _Float16* kn = qn + (size_t)NB * SL * HD;
    _Float16* vt = kn + (size_t)NB * SL * HD;

    cosattn_prep_kernel<<<dim3(NB * (SL / 64)), dim3(256), 0, stream>>>(q, k, v, qn, kn, vt);
    cosattn_main_kernel<<<dim3(NB * (SL / QT)), dim3(256), 0, stream>>>(qn, kn, vt, mask, out, attn);
}
